// LSTM_34720515621425
// MI455X (gfx1250) — matrix operand order repaired and checked by reading, not compiled, not run on hardware
//
#include <hip/hip_runtime.h>
#include <hip/hip_bf16.h>

#define Hh    512
#define Bb    32
#define Tt    1024
#define FOURH 2048

typedef __attribute__((ext_vector_type(16))) __bf16 v16bf;
typedef __attribute__((ext_vector_type(8)))  __bf16 v8bf;
typedef __attribute__((ext_vector_type(8)))  float  v8f;

typedef __attribute__((ext_vector_type(4))) unsigned int tdm_g0_t;
typedef __attribute__((ext_vector_type(8))) int          tdm_g1_t;
typedef __attribute__((ext_vector_type(4))) int          tdm_g4_t;

__device__ __forceinline__ __bf16 f2bf(float f) {
    unsigned u = __builtin_bit_cast(unsigned, f);
    unsigned r = (u + 0x7FFFu + ((u >> 16) & 1u)) >> 16;
    return __builtin_bit_cast(__bf16, (unsigned short)r);
}

__device__ __forceinline__ float sigmoidf(float x) {
    return 1.0f / (1.0f + __expf(-x));
}

__device__ __forceinline__ unsigned ldsoff(const void* p) {
    // generic LDS addresses carry the wave-relative LDS byte offset in the
    // low 32 bits (ISA 10.2 aperture mapping)
    return (unsigned)(unsigned long long)p;
}

// ---- Tensor Data Mover: 2D tile (tile_d1 rows x tile_d0 elems) -> LDS ----
// D# per ISA 8.3/8.4. Tiles are always fully in-bounds, so tensor dims are
// set equal to tile dims. Groups 2/3 zero => tile_dim2/3 unused (2D).
__device__ __forceinline__ void tdm_load_2d(unsigned lds_off, const void* gptr,
                                            unsigned data_size_code,
                                            unsigned tile_d0, unsigned tile_d1,
                                            unsigned long long stride_elems) {
    unsigned long long ga = (unsigned long long)gptr;
    tdm_g0_t g0;
    g0.x = 1u;                                    // count=1, user descriptor
    g0.y = lds_off;                               // lds_addr (bytes)
    g0.z = (unsigned)(ga & 0xFFFFFFFFu);          // global_addr[31:0]
    g0.w = (unsigned)((ga >> 32) & 0x1FFFFFFu)    // global_addr[56:32]
         | (2u << 30);                            // type = 2 ("image")
    tdm_g1_t g1;
    g1[0] = (int)(data_size_code << 16);          // wg_mask=0, data_size
    g1[1] = (int)((tile_d0 & 0xFFFFu) << 16);     // tensor_dim0[15:0]
    g1[2] = (int)(((tile_d0 >> 16) & 0xFFFFu)     // tensor_dim0[31:16]
         | ((tile_d1 & 0xFFFFu) << 16));          // tensor_dim1[15:0]
    g1[3] = (int)(((tile_d1 >> 16) & 0xFFFFu)     // tensor_dim1[31:16]
         | ((tile_d0 & 0xFFFFu) << 16));          // tile_dim0
    g1[4] = (int)(tile_d1 & 0xFFFFu);             // tile_dim1 (tile_dim2=0)
    g1[5] = (int)(stride_elems & 0xFFFFFFFFull);  // tensor_dim0_stride[31:0]
    g1[6] = (int)((stride_elems >> 32) & 0xFFFFull); // stride[47:32]
    g1[7] = 0;
    tdm_g4_t gz = {0, 0, 0, 0};
#if defined(__clang_major__) && (__clang_major__ >= 23)
    tdm_g1_t gz8 = {0, 0, 0, 0, 0, 0, 0, 0};
    __builtin_amdgcn_tensor_load_to_lds(g0, g1, gz, gz, gz8, 0);
#else
    __builtin_amdgcn_tensor_load_to_lds(g0, g1, gz, gz, 0);
#endif
}

// A fragment, 16x32 bf16 (MxK). Lanes 0-15: K 0-7 / 16-23; lanes 16-31:
// K 8-15 / 24-31. Row m = lane&15. Two 16B LDS reads.
__device__ __forceinline__ v16bf frag_a(const __bf16* a, int lda, int lane) {
    int m  = lane & 15;
    int kb = (lane >> 4) << 3;
    v8bf lo = *(const v8bf*)(a + m * lda + kb);
    v8bf hi = *(const v8bf*)(a + m * lda + 16 + kb);
    return __builtin_shufflevector(lo, hi, 0,1,2,3,4,5,6,7,8,9,10,11,12,13,14,15);
}

// B fragment, 32x16 bf16 (KxN), source stored N-major (row n holds K run).
__device__ __forceinline__ v16bf frag_b(const __bf16* b, int ldk, int lane) {
    int n  = lane & 15;
    int ko = (lane >> 4) << 3;
    v8bf lo = *(const v8bf*)(b + n * ldk + ko);
    v8bf hi = *(const v8bf*)(b + n * ldk + ko + 16);
    return __builtin_shufflevector(lo, hi, 0,1,2,3,4,5,6,7,8,9,10,11,12,13,14,15);
}

__global__ void k_cvt_bf16(const float* __restrict__ s, __bf16* __restrict__ d, int n) {
    int i = blockIdx.x * blockDim.x + threadIdx.x;
    if (i < n) d[i] = f2bf(s[i]);
}

__global__ void k_bias(const float* __restrict__ a, const float* __restrict__ b,
                       float* __restrict__ o, int n) {
    int i = blockIdx.x * blockDim.x + threadIdx.x;
    if (i < n) o[i] = a[i] + b[i];
}

__global__ void k_init(int* __restrict__ ctr, __bf16* __restrict__ h) {
    if (blockIdx.x == 0 && threadIdx.x == 0) ctr[0] = 0;
    for (int i = blockIdx.x * blockDim.x + threadIdx.x; i < 2 * Bb * Hh;
         i += blockDim.x * gridDim.x)
        h[i] = __builtin_bit_cast(__bf16, (unsigned short)0);
}

// gi[M, 4H] = A[M, 512] * W^T + bias   (A bf16 [M,512], W bf16 [4H,512])
// WG tile 64x64; 8 waves x 2 WMMA tiles (shared A frag). K blocked by 32,
// tiles staged into double-buffered LDS by the Tensor Data Mover; DMA of
// block k+1 overlaps WMMA of block k.
__global__ __launch_bounds__(256) void k_gemm_gi(const __bf16* __restrict__ A,
                                                 const __bf16* __restrict__ W,
                                                 const float* __restrict__ bias,
                                                 float* __restrict__ out) {
    __shared__ __align__(16) __bf16 a_lds[2][64 * 32];
    __shared__ __align__(16) __bf16 b_lds[2][64 * 32];
    int tid = threadIdx.x, lane = tid & 31, wave = tid >> 5;
    int m0w = blockIdx.x * 64, n0w = blockIdx.y * 64;
    int mt = wave >> 1, np = (wave & 1) << 1;

    unsigned a_off[2] = {ldsoff(a_lds[0]), ldsoff(a_lds[1])};
    unsigned b_off[2] = {ldsoff(b_lds[0]), ldsoff(b_lds[1])};

    v8f acc0 = {}, acc1 = {};
    if (wave == 0) {
        tdm_load_2d(a_off[0], A + (size_t)m0w * Hh, 1u, 32u, 64u, Hh);
        tdm_load_2d(b_off[0], W + (size_t)n0w * Hh, 1u, 32u, 64u, Hh);
        __builtin_amdgcn_s_wait_tensorcnt(0);
    }
    __syncthreads();

    for (int kk = 0; kk < Hh; kk += 32) {
        int buf = (kk >> 5) & 1;
        if (wave == 0 && (kk + 32) < Hh) {   // prefetch next K block
            tdm_load_2d(a_off[buf ^ 1], A + (size_t)m0w * Hh + kk + 32, 1u, 32u, 64u, Hh);
            tdm_load_2d(b_off[buf ^ 1], W + (size_t)n0w * Hh + kk + 32, 1u, 32u, 64u, Hh);
        }
        v16bf af  = frag_a(&a_lds[buf][mt * 16 * 32], 32, lane);
        v16bf bf0 = frag_b(&b_lds[buf][np * 16 * 32], 32, lane);
        v16bf bf1 = frag_b(&b_lds[buf][(np + 1) * 16 * 32], 32, lane);
        acc0 = __builtin_amdgcn_wmma_f32_16x16x32_bf16(false, af, false, bf0,
                                                       (short)0, acc0, false, false);
        acc1 = __builtin_amdgcn_wmma_f32_16x16x32_bf16(false, af, false, bf1,
                                                       (short)0, acc1, false, false);
        if (wave == 0) __builtin_amdgcn_s_wait_tensorcnt(0);
        __syncthreads();
    }

    int ngb = n0w + (np << 4) + (lane & 15);
    float bv0 = bias[ngb], bv1 = bias[ngb + 16];
    int mb = m0w + mt * 16 + ((lane >> 4) << 3);
    for (int r = 0; r < 8; ++r) {
        out[(size_t)(mb + r) * FOURH + ngb]      = acc0[r] + bv0;
        out[(size_t)(mb + r) * FOURH + ngb + 16] = acc1[r] + bv1;
    }
}

// Persistent recurrent scan. 32 WGs; WG owns 16 hidden units (all 4 gates).
// Whh slice (64 rows x 512) in LDS as bf16; c in registers; h double-buffered
// in global; per-step h broadcast staged into LDS by one TDM copy; grid
// barrier (monotonic counter) per step.
__global__ __launch_bounds__(256) void k_lstm_scan(
        const float* __restrict__ gi,   // [B, T, 4H] f32
        const float* __restrict__ whh,  // [4H, 512] f32
        __bf16* __restrict__ hbuf,      // [2][B*H] bf16
        __bf16* __restrict__ out_bf,    // [B,T,H] bf16 or null
        float*  __restrict__ out_f32,   // [B,T,H] f32 or null
        float* __restrict__ hn, float* __restrict__ cn,
        int* __restrict__ ctr) {
    __shared__ __align__(16) __bf16 w_lds[64 * Hh];   // 64 KB
    __shared__ __align__(16) __bf16 h_lds[Bb * Hh];   // 32 KB
    __shared__ float  g_lds[Bb * 64];                 // 8 KB

    int tid = threadIdx.x, lane = tid & 31, wave = tid >> 5;
    int j0 = blockIdx.x * 16;
    unsigned h_lds_off = ldsoff(h_lds);

    // stage Whh slice: local row lr = gate*16 + jj -> global row gate*512+j0+jj
    for (int i = tid; i < 64 * Hh; i += 256) {
        int lr = i >> 9, k = i & (Hh - 1);
        int gr = ((lr >> 4) * Hh) + j0 + (lr & 15);
        w_lds[i] = f2bf(whh[(size_t)gr * Hh + k]);
    }
    __syncthreads();

    int g  = wave & 3;
    int m0 = (wave >> 2) << 4;
    int ng = g * Hh + j0 + (lane & 15);

    int e0 = tid << 1, e1 = e0 + 1;
    int b0 = e0 >> 4, jj0 = e0 & 15;
    int b1 = e1 >> 4, jj1 = e1 & 15;
    float c0 = 0.f, c1 = 0.f;

    for (int t = 0; t < Tt; ++t) {
        const __bf16* hin = hbuf + (t & 1) * (Bb * Hh);
        __bf16* hout = hbuf + ((t + 1) & 1) * (Bb * Hh);

        // one TDM copy: 32 KB h state -> LDS (8-byte elements, 1 row)
        if (wave == 0) {
            tdm_load_2d(h_lds_off, hin, 3u, (Bb * Hh) / 4, 1u, (Bb * Hh) / 4);
            __builtin_amdgcn_s_wait_tensorcnt(0);
        }
        __syncthreads();

        v8f acc;
        for (int r = 0; r < 8; ++r) {
            int m = m0 + r + ((lane >> 4) << 3);
            acc[r] = gi[((size_t)m * Tt + t) * FOURH + ng];
        }
        for (int kk = 0; kk < Hh; kk += 32) {
            v16bf af = frag_a(h_lds + m0 * Hh + kk, Hh, lane);
            v16bf bf = frag_b(w_lds + (g * 16) * Hh + kk, Hh, lane);
            acc = __builtin_amdgcn_wmma_f32_16x16x32_bf16(false, af, false, bf,
                                                          (short)0, acc, false, false);
        }
        for (int r = 0; r < 8; ++r) {
            int m = m0 + r + ((lane >> 4) << 3);
            g_lds[m * 64 + g * 16 + (lane & 15)] = acc[r];
        }
        __syncthreads();

        {   // cell 0
            float iv = sigmoidf(g_lds[b0 * 64 +      jj0]);
            float fv = sigmoidf(g_lds[b0 * 64 + 16 + jj0]);
            float gv = tanhf   (g_lds[b0 * 64 + 32 + jj0]);
            float ov = sigmoidf(g_lds[b0 * 64 + 48 + jj0]);
            c0 = fv * c0 + iv * gv;
            float hv = ov * tanhf(c0);
            hout[b0 * Hh + j0 + jj0] = f2bf(hv);
            if (out_bf)  out_bf [((size_t)b0 * Tt + t) * Hh + j0 + jj0] = f2bf(hv);
            if (out_f32) out_f32[((size_t)b0 * Tt + t) * Hh + j0 + jj0] = hv;
            if (t == Tt - 1) { hn[b0 * Hh + j0 + jj0] = hv; cn[b0 * Hh + j0 + jj0] = c0; }
        }
        {   // cell 1
            float iv = sigmoidf(g_lds[b1 * 64 +      jj1]);
            float fv = sigmoidf(g_lds[b1 * 64 + 16 + jj1]);
            float gv = tanhf   (g_lds[b1 * 64 + 32 + jj1]);
            float ov = sigmoidf(g_lds[b1 * 64 + 48 + jj1]);
            c1 = fv * c1 + iv * gv;
            float hv = ov * tanhf(c1);
            hout[b1 * Hh + j0 + jj1] = f2bf(hv);
            if (out_bf)  out_bf [((size_t)b1 * Tt + t) * Hh + j0 + jj1] = f2bf(hv);
            if (out_f32) out_f32[((size_t)b1 * Tt + t) * Hh + j0 + jj1] = hv;
            if (t == Tt - 1) { hn[b1 * Hh + j0 + jj1] = hv; cn[b1 * Hh + j0 + jj1] = c1; }
        }

        if (t < Tt - 1) {
            __threadfence();
            __syncthreads();
            if (tid == 0) {
                atomicAdd(ctr, 1);
                int target = (t + 1) * (int)gridDim.x;
                while (__hip_atomic_load(ctr, __ATOMIC_ACQUIRE,
                                         __HIP_MEMORY_SCOPE_AGENT) < target)
                    __builtin_amdgcn_s_sleep(1);
            }
            __syncthreads();
            __threadfence();
        }
    }
}

extern "C" void kernel_launch(void* const* d_in, const int* in_sizes, int n_in,
                              void* d_out, int out_size, void* d_ws, size_t ws_size,
                              hipStream_t stream) {
    (void)in_sizes; (void)n_in; (void)out_size; (void)ws_size;
    const float* x    = (const float*)d_in[0];
    const float* Wih0 = (const float*)d_in[1];
    const float* Whh0 = (const float*)d_in[2];
    const float* bih0 = (const float*)d_in[3];
    const float* bhh0 = (const float*)d_in[4];
    const float* Wih1 = (const float*)d_in[5];
    const float* Whh1 = (const float*)d_in[6];
    const float* bih1 = (const float*)d_in[7];
    const float* bhh1 = (const float*)d_in[8];

    char* ws = (char*)d_ws;
    size_t off = 0;
    auto take = [&](size_t bytes) {
        char* p = ws + off;
        off = (off + bytes + 255) & ~(size_t)255;
        return p;
    };
    __bf16* xb   = (__bf16*)take((size_t)Bb * Tt * Hh * 2);
    __bf16* wbf  = (__bf16*)take((size_t)FOURH * Hh * 2);
    float*  bias = (float*) take((size_t)FOURH * 4);
    float*  gi   = (float*) take((size_t)Bb * Tt * FOURH * 4);
    __bf16* o0b  = (__bf16*)take((size_t)Bb * Tt * Hh * 2);
    __bf16* hbuf = (__bf16*)take((size_t)2 * Bb * Hh * 2);
    int*    ctr  = (int*)   take(256);

    float* out1 = (float*)d_out;
    float* hn   = out1 + (size_t)Bb * Tt * Hh;
    float* cn   = hn + 2 * (size_t)Bb * Hh;

    const int nx = Bb * Tt * Hh, nw = FOURH * Hh;
    dim3 gemm_grid(Bb * Tt / 64, FOURH / 64);

    // ---- layer 0 ----
    k_cvt_bf16<<<(nx + 255) / 256, 256, 0, stream>>>(x, xb, nx);
    k_cvt_bf16<<<(nw + 255) / 256, 256, 0, stream>>>(Wih0, wbf, nw);
    k_bias<<<(FOURH + 255) / 256, 256, 0, stream>>>(bih0, bhh0, bias, FOURH);
    k_gemm_gi<<<gemm_grid, 256, 0, stream>>>(xb, wbf, bias, gi);
    k_init<<<8, 256, 0, stream>>>(ctr, hbuf);
    k_lstm_scan<<<Hh / 16, 256, 0, stream>>>(gi, Whh0, hbuf, o0b, nullptr,
                                             hn, cn, ctr);

    // ---- layer 1 ----
    k_cvt_bf16<<<(nw + 255) / 256, 256, 0, stream>>>(Wih1, wbf, nw);
    k_bias<<<(FOURH + 255) / 256, 256, 0, stream>>>(bih1, bhh1, bias, FOURH);
    k_gemm_gi<<<gemm_grid, 256, 0, stream>>>(o0b, wbf, bias, gi);
    k_init<<<8, 256, 0, stream>>>(ctr, hbuf);
    k_lstm_scan<<<Hh / 16, 256, 0, stream>>>(gi, Whh1, hbuf, nullptr, out1,
                                             hn + Bb * Hh, cn + Bb * Hh, ctr);
}
